// MultiHeadAttention_48060684042459
// MI455X (gfx1250) — matrix operand order repaired and checked by reading, not compiled, not run on hardware
//
#include <hip/hip_runtime.h>

#define NSEQ   4096
#define DMODEL 1024
#define NHEADS 16
#define HDIM   64

typedef __bf16 bf16_t;
typedef __attribute__((ext_vector_type(16))) __bf16 v16bf;
typedef __attribute__((ext_vector_type(8)))  __bf16 v8bf;
typedef __attribute__((ext_vector_type(8)))  float  v8f;

static __device__ inline v8f vzero8() {
  v8f z;
#pragma unroll
  for (int i = 0; i < 8; ++i) z[i] = 0.0f;
  return z;
}

// D = A(16x32) * B(32x16) + C   (bf16 in, f32 accum)
static __device__ inline v8f wmma_bf16(v16bf a, v16bf b, v8f c) {
  return __builtin_amdgcn_wmma_f32_16x16x32_bf16(false, a, false, b, (short)0, c,
                                                 false, false);
}

// A-matrix per-lane fragment: two contiguous 8-element chunks at p and p+8.
// Caller pre-applies:  row = base + (lane&15),  k = k0 + (lane>=16 ? 16 : 0)
static __device__ inline v16bf load_a16(const bf16_t* p) {
  union { v16bf v; v8bf h[2]; } u;
  u.h[0] = *(const v8bf*)(p);
  u.h[1] = *(const v8bf*)(p + 8);
  return u.v;
}

// B-matrix per-lane fragment: 16 contiguous K-values of one column.
// Caller pre-applies:  col = base + (lane&15),  k = k0 + (lane>=16 ? 16 : 0)
static __device__ inline v16bf load_b16(const bf16_t* p) {
  return *(const v16bf*)p;
}

static __device__ inline unsigned pack_bf16(float lo, float hi) {
  union { unsigned u; __bf16 b[2]; } r;
  r.b[0] = (__bf16)lo;
  r.b[1] = (__bf16)hi;
  return r.u;
}

// opaque zero: defeats LICM/CSE on addresses without breaking the pointer's
// SSA chain (so address-space inference keeps loads as global_load, not flat)
static __device__ inline unsigned opaque_zero() {
  unsigned z = 0;
  asm volatile("" : "+v"(z));
  return z;
}

// ---------------------------------------------------------------- f32 -> bf16
__global__ void cvt_f32_bf16(const float* __restrict__ s, bf16_t* __restrict__ d,
                             int n) {
  int i = blockIdx.x * blockDim.x + threadIdx.x;
  if (i < n) d[i] = (bf16_t)s[i];
}

// ----------------------------------------------------- Y[M][N] = A[M][K] @ W[N][K]^T
// MODE 0: store bf16 row-major Y[M][N]
// MODE 1: store bf16 transposed Y^T[N][M]   (used for V -> vt[h*64+d][n])
// MODE 2: store f32 row-major with bias     (final output projection)
// One wave computes a 64(M) x 64(N) tile: 16 WMMAs per k-step on 16 b128 loads.
// blockDim = 256 (8 waves).
template <int MODE>
__global__ void gemm_bf16(const bf16_t* __restrict__ A, const bf16_t* __restrict__ W,
                          bf16_t* __restrict__ outB, float* __restrict__ outF,
                          const float* __restrict__ bias) {
  constexpr int M = NSEQ, N = DMODEL, K = DMODEL;
  const int lane = threadIdx.x & 31;
  const int half = lane >> 4;
  const int lr   = lane & 15;
  const int wave = (blockIdx.x * blockDim.x + threadIdx.x) >> 5;
  const int ntiles = N >> 6;               // 16
  const int mbase = (wave / ntiles) << 6;
  const int nbase = (wave % ntiles) << 6;

  v8f c[4][4];
#pragma unroll
  for (int i = 0; i < 4; ++i)
#pragma unroll
    for (int j = 0; j < 4; ++j) c[i][j] = vzero8();

  // loop-invariant per-row base pointers (k0 folds into the load immediates)
  const bf16_t* arow[4];
  const bf16_t* wrow[4];
#pragma unroll
  for (int i = 0; i < 4; ++i)
    arow[i] = A + (size_t)(mbase + i * 16 + lr) * K + (half ? 16 : 0);
#pragma unroll
  for (int j = 0; j < 4; ++j)
    wrow[j] = W + (size_t)(nbase + j * 16 + lr) * K + (half ? 16 : 0);

#pragma unroll 4
  for (int k0 = 0; k0 < K; k0 += 32) {
    __builtin_prefetch(arow[0] + k0 + 128, 0, 3);   // global_prefetch_b8
    // batch all fragments of this k-step so loads issue as one clause
    v16bf a[4], b[4];
#pragma unroll
    for (int i = 0; i < 4; ++i) a[i] = load_a16(arow[i] + k0);
#pragma unroll
    for (int j = 0; j < 4; ++j) b[j] = load_b16(wrow[j] + k0);
#pragma unroll
    for (int j = 0; j < 4; ++j)
#pragma unroll
      for (int i = 0; i < 4; ++i)
        c[i][j] = wmma_bf16(a[i], b[j], c[i][j]);
  }

#pragma unroll
  for (int j = 0; j < 4; ++j) {
    const int col = nbase + j * 16 + lr;
    const float bj = (MODE == 2) ? bias[col] : 0.0f;
#pragma unroll
    for (int i = 0; i < 4; ++i) {
#pragma unroll
      for (int v = 0; v < 8; ++v) {
        const int row = mbase + i * 16 + v + 8 * half;
        const float val = c[i][j][v];
        if (MODE == 0)      outB[(size_t)row * N + col] = (bf16_t)val;
        else if (MODE == 1) outB[(size_t)col * M + row] = (bf16_t)val;
        else                outF[(size_t)row * N + col] = val + bj;
      }
    }
  }
}

// ------------------------------------------------------------ flash attention
// One wave handles (head h, 32-query block qb), processed one 16-query tile at
// a time WITHIN each key-step so the live set stays ~130 VGPRs (no spills):
//   S^T = K @ Q^T     (query index == lane%16  -> per-lane softmax stats)
//   ctx^T = V^T @ P^T (accumulated in registers, 2 qtiles x 4 d-tiles x v8f)
// K / V^T / Q fragments are re-loaded per qtile via an opaque-zero offset
// (defeats LICM/CSE, keeps global addrspace): duplicate addresses are steady
// WGP$ hits, far cheaper than accumulator scratch-spills.
__global__ void flash_attn(const bf16_t* __restrict__ Q, const bf16_t* __restrict__ Km,
                           const bf16_t* __restrict__ Vt, bf16_t* __restrict__ Ctx) {
  const int qb   = blockIdx.x;      // 0..127
  const int h    = blockIdx.y;      // 0..15
  const int lane = threadIdx.x & 31;
  const int half = lane >> 4;
  const int lr   = lane & 15;
  const int qb32 = qb * 32;
  const int aoff  = half ? 16 : 0;
  const int boffq = half ? 16 : 0;
  const float CL2 = 0.1803368801111244f;  // log2(e) / sqrt(64)
  const float NEG_INF = -__builtin_inff();

  // loop-invariant base pointers (j0 folds into load offsets)
  const bf16_t* qp    = Q + (size_t)(qb32 + lr) * DMODEL + h * HDIM + boffq;
  const bf16_t* kbase = Km + (size_t)lr * DMODEL + h * HDIM + aoff;
  const bf16_t* vbase = Vt + (size_t)(h * HDIM + lr) * NSEQ + aoff;

  v8f acc[2][4];
#pragma unroll
  for (int qt = 0; qt < 2; ++qt)
#pragma unroll
    for (int dt = 0; dt < 4; ++dt) acc[qt][dt] = vzero8();
  float m[2]    = {NEG_INF, NEG_INF};
  float lsum[2] = {0.0f, 0.0f};

  union BP { v16bf v; unsigned u[8]; };

  for (int j0 = 0; j0 <= qb32 + 31; j0 += 32) {
#pragma unroll
    for (int qt = 0; qt < 2; ++qt) {
      const int qg = qb32 + qt * 16 + lr;   // this lane's query row
      const unsigned z0 = opaque_zero();    // per-qtile opaque offset

      // ---- Q as B-matrix fragments for this qtile (steady cache hits)
      const bf16_t* q0 = qp + (size_t)(qt * 16) * DMODEL + z0;
      v16bf bq0 = load_b16(q0);        // d 0..31
      v16bf bq1 = load_b16(q0 + 32);   // d 32..63

      // ---- batch all K fragments, then the 4 score WMMAs
      v16bf ka[2][2];  // [keytile][dstep]
#pragma unroll
      for (int t = 0; t < 2; ++t) {
        const bf16_t* krow = kbase + (size_t)(j0 + t * 16) * DMODEL + z0;
        ka[t][0] = load_a16(krow);        // d 0..31 chunk
        ka[t][1] = load_a16(krow + 32);   // d 32..63 chunk
      }
      v8f s[2];
#pragma unroll
      for (int t = 0; t < 2; ++t) {
        v8f z = vzero8();
        z = wmma_bf16(ka[t][0], bq0, z);
        z = wmma_bf16(ka[t][1], bq1, z);
        s[t] = z;
      }

      // ---- causal mask (only steps overlapping this qtile's diagonal)
      if (j0 + 31 > qb32 + qt * 16) {
#pragma unroll
        for (int t = 0; t < 2; ++t)
#pragma unroll
          for (int v = 0; v < 8; ++v) {
            const int kk = j0 + t * 16 + v + 8 * half;
            if (kk > qg) s[t][v] = NEG_INF;
          }
      }

      // ---- online softmax (per-lane stats == per query; pair-reduce xor 16)
      float mp = NEG_INF;
#pragma unroll
      for (int t = 0; t < 2; ++t)
#pragma unroll
        for (int v = 0; v < 8; ++v) mp = fmaxf(mp, s[t][v]);
      mp = fmaxf(mp, __shfl_xor(mp, 16, 32));
      const float mnew  = fmaxf(m[qt], mp);
      const float alpha = exp2f((m[qt] - mnew) * CL2);
      float ssum = 0.0f;
#pragma unroll
      for (int t = 0; t < 2; ++t)
#pragma unroll
        for (int v = 0; v < 8; ++v) {
          const float p = exp2f((s[t][v] - mnew) * CL2);
          s[t][v] = p;
          ssum += p;
        }
      ssum += __shfl_xor(ssum, 16, 32);
      lsum[qt] = lsum[qt] * alpha + ssum;
      m[qt] = mnew;
#pragma unroll
      for (int dt = 0; dt < 4; ++dt)
#pragma unroll
        for (int v = 0; v < 8; ++v) acc[qt][dt][v] *= alpha;

      // ---- P^T into B-layout: exchange key-halves between lane pairs (l, l^16)
      unsigned t0[4], t1[4];
#pragma unroll
      for (int jj = 0; jj < 4; ++jj) {
        t0[jj] = pack_bf16(s[0][2 * jj], s[0][2 * jj + 1]);
        t1[jj] = pack_bf16(s[1][2 * jj], s[1][2 * jj + 1]);
      }
      BP bp;
#pragma unroll
      for (int jj = 0; jj < 4; ++jj) {
        const unsigned e0 = (unsigned)__shfl_xor((int)t0[jj], 16, 32);
        const unsigned e1 = (unsigned)__shfl_xor((int)t1[jj], 16, 32);
        bp.u[jj]     = half ? e1 : t0[jj];   // K 0..7   | 16..23
        bp.u[4 + jj] = half ? t1[jj] : e0;   // K 8..15  | 24..31
      }

      // ---- batch all V^T fragments, then the 4 ctx WMMAs
      v16bf av[4];
#pragma unroll
      for (int dt = 0; dt < 4; ++dt)
        av[dt] = load_a16(vbase + (size_t)(dt * 16) * NSEQ + j0 + z0);
#pragma unroll
      for (int dt = 0; dt < 4; ++dt)
        acc[qt][dt] = wmma_bf16(av[dt], bp.v, acc[qt][dt]);
    }
  }

  // ---- normalize and store ctx row-major bf16 (d = dt*16 + v + 8*half, q = lr)
#pragma unroll
  for (int qt = 0; qt < 2; ++qt) {
    const float inv = 1.0f / lsum[qt];
    bf16_t* crow = Ctx + (size_t)(qb32 + qt * 16 + lr) * DMODEL + h * HDIM;
#pragma unroll
    for (int dt = 0; dt < 4; ++dt)
#pragma unroll
      for (int v = 0; v < 8; ++v)
        crow[dt * 16 + v + 8 * half] = (bf16_t)(acc[qt][dt][v] * inv);
  }
}

// ---------------------------------------------------------------------------
extern "C" void kernel_launch(void* const* d_in, const int* in_sizes, int n_in,
                              void* d_out, int out_size, void* d_ws, size_t ws_size,
                              hipStream_t stream) {
  (void)in_sizes; (void)n_in; (void)out_size; (void)ws_size;
  const float* x  = (const float*)d_in[0];
  const float* Wq = (const float*)d_in[1];
  const float* Wk = (const float*)d_in[2];
  const float* Wv = (const float*)d_in[3];
  const float* Wo = (const float*)d_in[4];
  const float* bo = (const float*)d_in[5];
  float* out = (float*)d_out;

  char* ws = (char*)d_ws;
  const size_t MB = (size_t)1 << 20;
  bf16_t* xb  = (bf16_t*)(ws + 0 * MB);    // [4096][1024]      8 MB
  bf16_t* wqb = (bf16_t*)(ws + 8 * MB);    // [1024][1024]      2 MB
  bf16_t* wkb = (bf16_t*)(ws + 10 * MB);
  bf16_t* wvb = (bf16_t*)(ws + 12 * MB);
  bf16_t* wob = (bf16_t*)(ws + 14 * MB);
  bf16_t* qbm = (bf16_t*)(ws + 16 * MB);   // Q  [4096][1024]   8 MB
  bf16_t* kbm = (bf16_t*)(ws + 24 * MB);   // K  [4096][1024]   8 MB
  bf16_t* vtm = (bf16_t*)(ws + 32 * MB);   // V^T [1024][4096]  8 MB
  bf16_t* cxb = (bf16_t*)(ws + 40 * MB);   // ctx [4096][1024]  8 MB

  const int nx = NSEQ * DMODEL;
  const int nw = DMODEL * DMODEL;
  cvt_f32_bf16<<<(nx + 255) / 256, 256, 0, stream>>>(x, xb, nx);
  cvt_f32_bf16<<<(nw + 255) / 256, 256, 0, stream>>>(Wq, wqb, nw);
  cvt_f32_bf16<<<(nw + 255) / 256, 256, 0, stream>>>(Wk, wkb, nw);
  cvt_f32_bf16<<<(nw + 255) / 256, 256, 0, stream>>>(Wv, wvb, nw);
  cvt_f32_bf16<<<(nw + 255) / 256, 256, 0, stream>>>(Wo, wob, nw);

  // (4096/64)*(1024/64) waves / 8 waves per 256-thread block = 128 blocks
  const int gblocks = (NSEQ / 64) * (DMODEL / 64) / 8;
  gemm_bf16<0><<<gblocks, 256, 0, stream>>>(xb, wqb, qbm, nullptr, nullptr);
  gemm_bf16<0><<<gblocks, 256, 0, stream>>>(xb, wkb, kbm, nullptr, nullptr);
  gemm_bf16<1><<<gblocks, 256, 0, stream>>>(xb, wvb, vtm, nullptr, nullptr);

  flash_attn<<<dim3(NSEQ / 32, NHEADS), 32, 0, stream>>>(qbm, kbm, vtm, cxb);

  gemm_bf16<2><<<gblocks, 256, 0, stream>>>(cxb, wob, nullptr, out, bo);
}
